// PrototypeAwareExplanation_19335942767228
// MI455X (gfx1250) — matrix operand order repaired and checked by reading, not compiled, not run on hardware
//
#include <hip/hip_runtime.h>
#include <hip/hip_bf16.h>
#include <cstdint>
#include <cstddef>

// ---------------------------------------------------------------------------
// CDNA5 (gfx1250) wave32 WMMA implementation of PrototypeAwareExplanation.
// GEMMs run in bf16 via v_wmma_f32_16x16x32_bf16; bf16 activation tiles are
// staged into LDS with global_load_async_to_lds_b128 (ASYNCcnt path) or the
// Tensor Data Mover (tensor_load_to_lds, TENSORcnt path, LDS pad feature).
// ---------------------------------------------------------------------------

typedef __bf16 bf16_t;
typedef bf16_t v16bf  __attribute__((ext_vector_type(16)));
typedef bf16_t bf16x2 __attribute__((ext_vector_type(2)));
typedef float  v8f    __attribute__((ext_vector_type(8)));
typedef unsigned int v4u __attribute__((ext_vector_type(4)));
typedef int          v4i __attribute__((ext_vector_type(4)));
typedef int          v8i __attribute__((ext_vector_type(8)));

#define HD   512      // hidden dim
#define MD   128      // metadata dim
#define NP   12       // prototypes
#define TOPK 4
#define RSCALE 0.2f

static __device__ __forceinline__ bf16_t f2bf(float x) { return (bf16_t)x; }
static __device__ __forceinline__ float  bf2f(bf16_t x) { return (float)x; }

static __device__ __forceinline__ float gelu_exact(float x) {
    return 0.5f * x * (1.0f + erff(x * 0.7071067811865475f));
}
static __device__ __forceinline__ float sigmoidf(float x) {
    return 1.0f / (1.0f + expf(-x));
}
static __device__ __forceinline__ float wave_allreduce_add(float v) {
    #pragma unroll
    for (int m = 16; m >= 1; m >>= 1) v += __shfl_xor(v, m, 32);
    return v;
}
static __device__ __forceinline__ int clampi(int v, int lo, int hi) {
    return v < lo ? lo : (v > hi ? hi : v);
}

// ---------------------------------------------------------------------------
// f32 -> bf16 weight pre-conversion (weights then live in L2 as bf16)
// ---------------------------------------------------------------------------
__global__ __launch_bounds__(256) void cvt_bf16_kernel(
    const float* __restrict__ src, bf16_t* __restrict__ dst, int n)
{
    int i0 = (blockIdx.x * 256 + threadIdx.x) * 4;
    if (i0 >= n) return;
    float4 v = *(const float4*)(src + i0);
    dst[i0 + 0] = f2bf(v.x);
    dst[i0 + 1] = f2bf(v.y);
    dst[i0 + 2] = f2bf(v.z);
    dst[i0 + 3] = f2bf(v.w);
}

// ---------------------------------------------------------------------------
// Tiled GEMM:  out[N, 512] = A[N, K] @ B[K, 512] + bias
//   - workgroup tile: 128 rows x 128 cols, 8 waves, each wave 32x64
//   - K stepped by 32; bf16 operands staged in padded LDS (KP=40: 80B rows,
//     16B aligned, 20-bank step -> conflict-free fragment reads)
//   - AMODE 0: A = r_in (constructed; segment uniform per K-block)
//     AMODE 1: A = feats (constructed; selector uniform per K-block)
//     AMODE 2: A = bf16, staged via global_load_async_to_lds_b128 (ASYNCcnt)
//     AMODE 3: A = bf16, staged via tensor_load_to_lds TDM with LDS padding
//              (pad_interval=3 -> every 16 DWORDs, pad_amount=3 -> 4 DWORDs,
//               so LDS row stride = 64+16 = 80B = KP) (TENSORcnt)
// ---------------------------------------------------------------------------
template <int AMODE>
__global__ __launch_bounds__(256) void gemm_tile_kernel(
    const float* __restrict__ pairE, const float* __restrict__ ep,
    const int* __restrict__ sid, const int* __restrict__ hc, const int* __restrict__ ps,
    const float* __restrict__ schemaE, const float* __restrict__ hopE,
    const float* __restrict__ srcE,
    const float* __restrict__ ctx,            // AMODE 1
    const bf16_t* __restrict__ Abf,           // AMODE 2/3
    const bf16_t* __restrict__ Bbf,           // [K, 512] row-major bf16
    const float* __restrict__ bias,           // [512]
    float* __restrict__ outp,                 // [N, 512] f32
    int K, int Nrows)
{
    constexpr int KP = 40;                     // padded K stride (elements)
    __shared__ bf16_t lA[128 * KP];
    __shared__ bf16_t lB[128 * KP];

    const int tid  = threadIdx.x;
    const int lane = tid & 31;
    const int wave = tid >> 5;
    const int wm   = wave & 3;                 // 32-row band within tile
    const int wn   = wave >> 2;                // 64-col band within tile
    const int row0 = blockIdx.x * 128;
    const int col0 = blockIdx.y * 128;

    v8f acc[2][4];
    #pragma unroll
    for (int mi = 0; mi < 2; ++mi)
        #pragma unroll
        for (int ni = 0; ni < 4; ++ni)
            #pragma unroll
            for (int e = 0; e < 8; ++e) acc[mi][ni][e] = 0.0f;

    const int mloc = lane & 15;
    const int kbA  = (lane & 16) ? 16 : 0;     // A fragment K split (contiguous, matches B)
    const int kbB  = (lane & 16) ? 16 : 0;     // B fragment K split

    const uint32_t ldsA = (uint32_t)(size_t)(&lA[0]);

    for (int k0 = 0; k0 < K; k0 += 32) {
        __syncthreads();

        // ---- stage A tile (128 x 32) as bf16 ----
        if constexpr (AMODE == 2) {
            // async DMA: 512 x 16-byte chunks, per-lane LDS scatter keeps pad
            #pragma unroll
            for (int i = 0; i < 2; ++i) {
                int c = tid + i * 256;                 // 0..511
                int m = c >> 2, q = c & 3;             // row, 16B quarter
                uint64_t ga = (uint64_t)(const void*)(Abf)
                            + (uint64_t)(row0 + m) * (HD * 2) + (uint32_t)k0 * 2 + q * 16;
                uint32_t la = ldsA + (uint32_t)(m * KP) * 2 + q * 16;
                asm volatile("global_load_async_to_lds_b128 %0, %1, off"
                             :: "v"(la), "v"(ga) : "memory");
            }
        } else if constexpr (AMODE == 3) {
            // Tensor Data Mover: one 2D tile descriptor per K-step, wave 0 only
            if (wave == 0) {
                uint64_t ga = (uint64_t)(const void*)(Abf)
                            + (uint64_t)row0 * (HD * 2) + (uint32_t)k0 * 2;
                v4u g0;
                g0[0] = 1u;                            // count=1, user mode
                g0[1] = ldsA;                          // lds_addr (bytes)
                g0[2] = (uint32_t)ga;                  // global_addr[31:0]
                g0[3] = (uint32_t)((ga >> 32) & 0x01FFFFFFu) | 0x80000000u; // [56:32] | type=2
                v8i g1;
                g1[0] = (1 << 16)                      // data_size = 2 bytes
                      | (1 << 20)                      // pad_enable
                      | (3 << 22)                      // pad_interval: 16 DWORDs
                      | (3 << 25);                     // pad_amount: 4 DWORDs
                g1[1] = (HD << 16);                    // tensor_dim0 = 512 (bits 79:48)
                g1[2] = (int)((unsigned)(Nrows & 0xFFFF) << 16);  // tensor_dim1[15:0]
                g1[3] = (Nrows >> 16) | (32 << 16);    // tensor_dim1[31:16] | tile_dim0=32
                g1[4] = 128;                           // tile_dim1=128, tile_dim2=0
                g1[5] = HD;                            // tensor_dim0_stride = 512
                g1[6] = 0;
                g1[7] = 0;
                v4i gz4 = {0, 0, 0, 0};
#if __clang_major__ >= 23
                v8i gz8 = {0, 0, 0, 0, 0, 0, 0, 0};
                __builtin_amdgcn_tensor_load_to_lds(g0, g1, gz4, gz4, gz8, 0);
#else
                __builtin_amdgcn_tensor_load_to_lds(g0, g1, gz4, gz4, 0);
#endif
                __builtin_amdgcn_s_wait_tensorcnt(0);
            }
        } else if constexpr (AMODE == 0) {
            if (k0 < 1024) {                           // pair | ep segments
                const float* base = (k0 < 512) ? pairE : ep;
                const int off = k0 & 511;
                #pragma unroll
                for (int i = 0; i < 16; ++i) {
                    int idx = tid + i * 256;
                    int m = idx >> 5, kk = idx & 31;
                    lA[m * KP + kk] = f2bf(base[(size_t)(row0 + m) * HD + off + kk]);
                }
            } else {                                   // metadata gather segments
                const float* tbl; const int* ids; int hi, off;
                if (k0 < 1152)      { tbl = schemaE; ids = sid; hi = 2048; off = k0 - 1024; }
                else if (k0 < 1280) { tbl = hopE;    ids = hc;  hi = 9;    off = k0 - 1152; }
                else                { tbl = srcE;    ids = ps;  hi = 7;    off = k0 - 1280; }
                #pragma unroll
                for (int i = 0; i < 16; ++i) {
                    int idx = tid + i * 256;
                    int m = idx >> 5, kk = idx & 31;
                    int id = clampi(ids[row0 + m], 0, hi);
                    lA[m * KP + kk] = f2bf(tbl[(size_t)id * MD + off + kk]);
                }
            }
        } else {                                       // AMODE 1: feats
            const int sel = k0 >> 9;
            const int off = k0 & 511;
            #pragma unroll
            for (int i = 0; i < 16; ++i) {
                int idx = tid + i * 256;
                int m = idx >> 5, kk = idx & 31;
                size_t p = (size_t)(row0 + m) * HD + off + kk;
                float v;
                if (sel == 0)      v = ep[p];
                else if (sel == 1) v = ctx[p];
                else if (sel == 2) v = fabsf(ep[p] - ctx[p]);
                else               v = ep[p] * ctx[p];
                lA[m * KP + kk] = f2bf(v);
            }
        }

        // ---- stage B tile (32 x 128), transposed to [n][k] for frag loads ----
        #pragma unroll
        for (int i = 0; i < 8; ++i) {
            int idx = tid + i * 256;                   // 2048 dword copies
            int kk = idx >> 6, nn = (idx & 63) * 2;    // coalesced along n
            bf16x2 p = *(const bf16x2*)(Bbf + (size_t)(k0 + kk) * HD + col0 + nn);
            lB[nn * KP + kk]       = p[0];
            lB[(nn + 1) * KP + kk] = p[1];
        }

        // prefetch next B tile into caches
        if (k0 + 32 < K) {
            const bf16_t* nb = Bbf + (size_t)(k0 + 32 + (tid >> 3)) * HD
                                   + col0 + (tid & 7) * 16;
            __builtin_prefetch(nb, 0, 2);
        }

        if constexpr (AMODE == 2) {
            asm volatile("s_wait_asynccnt 0x0" ::: "memory");
        }
        __syncthreads();

        // ---- load fragments (CDNA5 16-bit A/B VGPR layouts) ----
        v16bf af[2], bfv[4];
        #pragma unroll
        for (int mi = 0; mi < 2; ++mi) {
            const bf16_t* p = lA + (wm * 32 + mi * 16 + mloc) * KP;
            #pragma unroll
            for (int e = 0; e < 8; ++e) af[mi][e] = p[kbA + e];
            #pragma unroll
            for (int e = 0; e < 8; ++e) af[mi][8 + e] = p[kbA + 8 + e];
        }
        #pragma unroll
        for (int ni = 0; ni < 4; ++ni) {
            const bf16_t* p = lB + (wn * 64 + ni * 16 + mloc) * KP + kbB;
            #pragma unroll
            for (int e = 0; e < 16; ++e) bfv[ni][e] = p[e];
        }
        // ---- 8 WMMAs per K-step per wave ----
        #pragma unroll
        for (int mi = 0; mi < 2; ++mi)
            #pragma unroll
            for (int ni = 0; ni < 4; ++ni)
                acc[mi][ni] = __builtin_amdgcn_wmma_f32_16x16x32_bf16(
                    false, af[mi], false, bfv[ni], (short)0, acc[mi][ni], false, false);
    }

    // ---- epilogue: C layout (VGPR r -> row r + 8*(lane>=16), col = lane&15) ----
    const int rhi = (lane & 16) ? 8 : 0;
    #pragma unroll
    for (int mi = 0; mi < 2; ++mi) {
        #pragma unroll
        for (int ni = 0; ni < 4; ++ni) {
            int col = col0 + wn * 64 + ni * 16 + (lane & 15);
            float bv = bias[col];
            #pragma unroll
            for (int r = 0; r < 8; ++r) {
                int row = row0 + wm * 32 + mi * 16 + rhi + r;
                outp[(size_t)row * HD + col] = acc[mi][ni][r] + bv;
            }
        }
    }
}

// ---------------------------------------------------------------------------
// LN -> GELU -> bf16   (one wave per row, 16 cols per lane)
// ---------------------------------------------------------------------------
__global__ __launch_bounds__(256) void ln_gelu_kernel(
    const float* __restrict__ pre, const float* __restrict__ g,
    const float* __restrict__ b, bf16_t* __restrict__ outp)
{
    const int lane = threadIdx.x & 31;
    const int row  = blockIdx.x * 8 + (threadIdx.x >> 5);
    const float* x = pre + (size_t)row * HD + lane * 16;
    float v[16], s = 0.f, s2 = 0.f;
    #pragma unroll
    for (int j = 0; j < 16; ++j) { v[j] = x[j]; s += v[j]; s2 += v[j] * v[j]; }
    s = wave_allreduce_add(s); s2 = wave_allreduce_add(s2);
    float mu = s * (1.0f / HD);
    float rstd = rsqrtf(s2 * (1.0f / HD) - mu * mu + 1e-5f);
    #pragma unroll
    for (int j = 0; j < 16; ++j) {
        int c = lane * 16 + j;
        float y = (v[j] - mu) * rstd * g[c] + b[c];
        outp[(size_t)row * HD + c] = f2bf(gelu_exact(y));
    }
}

// ---------------------------------------------------------------------------
// Router tail: logits -> softmax -> top-4 renorm -> weights out, ctx out
// ---------------------------------------------------------------------------
__global__ __launch_bounds__(256) void router_tail_kernel(
    const bf16_t* __restrict__ h, const float* __restrict__ w2,
    const float* __restrict__ b2, const float* __restrict__ proto,
    float* __restrict__ outCtx, float* __restrict__ outW)
{
    __shared__ float sW2[HD * NP];   // 24KB
    __shared__ float sP[NP * HD];    // 24KB
    const int tid = threadIdx.x;
    #pragma unroll
    for (int i = 0; i < 24; ++i) {
        int idx = tid + i * 256;
        sW2[idx] = w2[idx];
        sP[idx]  = proto[idx];
    }
    __syncthreads();

    const int lane = tid & 31;
    const int row  = blockIdx.x * 8 + (tid >> 5);
    const bf16_t* hrow = h + (size_t)row * HD;

    float lg[NP];
    #pragma unroll
    for (int p = 0; p < NP; ++p) lg[p] = 0.f;
    #pragma unroll
    for (int j = 0; j < 16; ++j) {
        int k = lane * 16 + j;
        float hv = bf2f(hrow[k]);
        const float* wr = sW2 + k * NP;
        #pragma unroll
        for (int p = 0; p < NP; ++p) lg[p] += hv * wr[p];
    }
    #pragma unroll
    for (int p = 0; p < NP; ++p) lg[p] = wave_allreduce_add(lg[p]) + b2[p];

    // softmax (TEMP = 1)
    float mx = lg[0];
    #pragma unroll
    for (int p = 1; p < NP; ++p) mx = fmaxf(mx, lg[p]);
    float wv[NP], ssum = 0.f;
    #pragma unroll
    for (int p = 0; p < NP; ++p) { wv[p] = expf(lg[p] - mx); ssum += wv[p]; }
    float inv = 1.0f / ssum;
    #pragma unroll
    for (int p = 0; p < NP; ++p) wv[p] *= inv;

    // top-4 (first-occurrence on ties, like top_k) + renormalize
    int selmask = 0;
    float s4 = 0.f;
    for (int t = 0; t < TOPK; ++t) {
        float best = -1.f; int bi = 0;
        #pragma unroll
        for (int p = 0; p < NP; ++p)
            if (!((selmask >> p) & 1) && wv[p] > best) { best = wv[p]; bi = p; }
        selmask |= (1 << bi);
        s4 += best;
    }
    float dinv = 1.0f / fmaxf(s4, 1e-6f);
    float wfin[NP];
    #pragma unroll
    for (int p = 0; p < NP; ++p)
        wfin[p] = ((selmask >> p) & 1) ? wv[p] * dinv : 0.f;

    // ctx = wfin @ prototype_bank
    #pragma unroll
    for (int j = 0; j < 16; ++j) {
        int c = lane * 16 + j;
        float a = 0.f;
        #pragma unroll
        for (int p = 0; p < NP; ++p) a += wfin[p] * sP[p * HD + c];
        outCtx[(size_t)row * HD + c] = a;
    }
    if (lane < NP) outW[(size_t)row * NP + lane] = wfin[lane];
}

// ---------------------------------------------------------------------------
// t1 = bf16(GELU(LN(pre1)))  ;  t2 = bf16(GELU(pre2))
// ---------------------------------------------------------------------------
__global__ __launch_bounds__(256) void act_pair_kernel(
    const float* __restrict__ pre1, const float* __restrict__ g,
    const float* __restrict__ b, const float* __restrict__ pre2,
    bf16_t* __restrict__ t1, bf16_t* __restrict__ t2)
{
    const int lane = threadIdx.x & 31;
    const int row  = blockIdx.x * 8 + (threadIdx.x >> 5);
    const float* x1 = pre1 + (size_t)row * HD + lane * 16;
    const float* x2 = pre2 + (size_t)row * HD + lane * 16;
    float v[16], s = 0.f, s2 = 0.f;
    #pragma unroll
    for (int j = 0; j < 16; ++j) { v[j] = x1[j]; s += v[j]; s2 += v[j] * v[j]; }
    s = wave_allreduce_add(s); s2 = wave_allreduce_add(s2);
    float mu = s * (1.0f / HD);
    float rstd = rsqrtf(s2 * (1.0f / HD) - mu * mu + 1e-5f);
    #pragma unroll
    for (int j = 0; j < 16; ++j) {
        int c = lane * 16 + j;
        float y = (v[j] - mu) * rstd * g[c] + b[c];
        t1[(size_t)row * HD + c] = f2bf(gelu_exact(y));
        t2[(size_t)row * HD + c] = f2bf(gelu_exact(x2[j]));
    }
}

// ---------------------------------------------------------------------------
// updated = LN(ep + RSCALE * sigmoid(gate_pre) * delta_pre) * g + b
// ---------------------------------------------------------------------------
__global__ __launch_bounds__(256) void final_kernel(
    const float* __restrict__ ep, const float* __restrict__ dpre,
    const float* __restrict__ gpre, const float* __restrict__ g,
    const float* __restrict__ b, float* __restrict__ outU)
{
    const int lane = threadIdx.x & 31;
    const int row  = blockIdx.x * 8 + (threadIdx.x >> 5);
    const size_t base = (size_t)row * HD + lane * 16;
    float u[16], s = 0.f, s2 = 0.f;
    #pragma unroll
    for (int j = 0; j < 16; ++j) {
        u[j] = ep[base + j] + RSCALE * sigmoidf(gpre[base + j]) * dpre[base + j];
        s += u[j]; s2 += u[j] * u[j];
    }
    s = wave_allreduce_add(s); s2 = wave_allreduce_add(s2);
    float mu = s * (1.0f / HD);
    float rstd = rsqrtf(s2 * (1.0f / HD) - mu * mu + 1e-5f);
    #pragma unroll
    for (int j = 0; j < 16; ++j) {
        int c = lane * 16 + j;
        outU[(size_t)row * HD + c] = (u[j] - mu) * rstd * g[c] + b[c];
    }
}

// ---------------------------------------------------------------------------
// Host-side orchestration
// ---------------------------------------------------------------------------
extern "C" void kernel_launch(void* const* d_in, const int* in_sizes, int n_in,
                              void* d_out, int out_size, void* d_ws, size_t ws_size,
                              hipStream_t stream) {
    const float* pairE   = (const float*)d_in[0];
    const float* ep      = (const float*)d_in[1];
    const int*   sid     = (const int*)  d_in[2];
    const int*   hc      = (const int*)  d_in[3];
    const int*   ps      = (const int*)  d_in[4];
    const float* schemaE = (const float*)d_in[5];
    const float* hopE    = (const float*)d_in[6];
    const float* srcE    = (const float*)d_in[7];
    const float* proto   = (const float*)d_in[8];
    const float* rw1     = (const float*)d_in[9];
    const float* rb1     = (const float*)d_in[10];
    const float* rlng    = (const float*)d_in[11];
    const float* rlnb    = (const float*)d_in[12];
    const float* rw2     = (const float*)d_in[13];
    const float* rb2     = (const float*)d_in[14];
    const float* muw1    = (const float*)d_in[15];
    const float* mub1    = (const float*)d_in[16];
    const float* mulng   = (const float*)d_in[17];
    const float* mulnb   = (const float*)d_in[18];
    const float* muw2    = (const float*)d_in[19];
    const float* mub2    = (const float*)d_in[20];
    const float* mgw1    = (const float*)d_in[21];
    const float* mgb1    = (const float*)d_in[22];
    const float* mgw2    = (const float*)d_in[23];
    const float* mgb2    = (const float*)d_in[24];
    const float* normg   = (const float*)d_in[25];
    const float* normb   = (const float*)d_in[26];

    const int N = in_sizes[0] / HD;            // 65536
    const size_t nh = (size_t)N * HD;

    // workspace: pre1 f32 | pre2 f32 | h bf16 | t1 bf16 | t2 bf16 | bf16 weights
    char* wsb = (char*)d_ws;
    float*  pre1 = (float*)(wsb);
    float*  pre2 = (float*)(wsb + nh * 4);
    bf16_t* hbuf = (bf16_t*)(wsb + nh * 8);
    bf16_t* t1   = (bf16_t*)(wsb + nh * 10);
    bf16_t* t2   = (bf16_t*)(wsb + nh * 12);
    bf16_t* rw1b  = (bf16_t*)(wsb + nh * 14);
    bf16_t* muw1b = rw1b  + (size_t)1408 * HD;
    bf16_t* mgw1b = muw1b + (size_t)2048 * HD;
    bf16_t* muw2b = mgw1b + (size_t)2048 * HD;
    bf16_t* mgw2b = muw2b + (size_t)HD * HD;

    float* outU = (float*)d_out;               // [N,512]
    float* outC = outU + nh;                   // [N,512]
    float* outW = outC + nh;                   // [N,12]

    dim3 gGemm(N / 128, HD / 128);             // (512, 4)
    dim3 gRow(N / 8);                          // wave per row, 8 waves/block
    dim3 blk(256);

    // 0) weights -> bf16 (stay hot in the 192MB L2 across all GEMMs)
    cvt_bf16_kernel<<<dim3(1408 * HD / 1024), blk, 0, stream>>>(rw1,  rw1b,  1408 * HD);
    cvt_bf16_kernel<<<dim3(2048 * HD / 1024), blk, 0, stream>>>(muw1, muw1b, 2048 * HD);
    cvt_bf16_kernel<<<dim3(2048 * HD / 1024), blk, 0, stream>>>(mgw1, mgw1b, 2048 * HD);
    cvt_bf16_kernel<<<dim3(HD * HD / 1024),   blk, 0, stream>>>(muw2, muw2b, HD * HD);
    cvt_bf16_kernel<<<dim3(HD * HD / 1024),   blk, 0, stream>>>(mgw2, mgw2b, HD * HD);

    // 1) router GEMM: r_in @ router_w1 + b1
    gemm_tile_kernel<0><<<gGemm, blk, 0, stream>>>(
        pairE, ep, sid, hc, ps, schemaE, hopE, srcE,
        nullptr, nullptr, rw1b, rb1, pre1, 2 * HD + 3 * MD, N);
    // 2) h = bf16(GELU(LN(pre1)))
    ln_gelu_kernel<<<gRow, blk, 0, stream>>>(pre1, rlng, rlnb, hbuf);
    // 3) logits/softmax/top4 -> weights, ctx
    router_tail_kernel<<<gRow, blk, 0, stream>>>(hbuf, rw2, rb2, proto, outC, outW);
    // 4) feats @ mu_w1 + b ; feats @ mg_w1 + b
    gemm_tile_kernel<1><<<gGemm, blk, 0, stream>>>(
        nullptr, ep, nullptr, nullptr, nullptr, nullptr, nullptr, nullptr,
        outC, nullptr, muw1b, mub1, pre1, 4 * HD, N);
    gemm_tile_kernel<1><<<gGemm, blk, 0, stream>>>(
        nullptr, ep, nullptr, nullptr, nullptr, nullptr, nullptr, nullptr,
        outC, nullptr, mgw1b, mgb1, pre2, 4 * HD, N);
    // 5) t1 = GELU(LN(pre1)); t2 = GELU(pre2)
    act_pair_kernel<<<gRow, blk, 0, stream>>>(pre1, mulng, mulnb, pre2, t1, t2);
    // 6) delta_pre = t1 @ mu_w2 + b (async-LDS A staging)
    gemm_tile_kernel<2><<<gGemm, blk, 0, stream>>>(
        nullptr, nullptr, nullptr, nullptr, nullptr, nullptr, nullptr, nullptr,
        nullptr, t1, muw2b, mub2, pre1, HD, N);
    //    gate_pre = t2 @ mg_w2 + b (Tensor Data Mover A staging)
    gemm_tile_kernel<3><<<gGemm, blk, 0, stream>>>(
        nullptr, nullptr, nullptr, nullptr, nullptr, nullptr, nullptr, nullptr,
        nullptr, t2, mgw2b, mgb2, pre2, HD, N);
    // 7) updated = LN(ep + 0.2 * sigmoid(gate) * delta)
    final_kernel<<<gRow, blk, 0, stream>>>(ep, pre1, pre2, normg, normb, outU);
}
